// MultiHeadAttention_54125177864217
// MI455X (gfx1250) — matrix operand order repaired and checked by reading, not compiled, not run on hardware
//
#include <hip/hip_runtime.h>
#include <hip/hip_bf16.h>
#include <stdint.h>

typedef __bf16 bf16;
typedef __attribute__((ext_vector_type(16))) __bf16 v16bf;
typedef __attribute__((ext_vector_type(8)))  __bf16 v8bf;
typedef __attribute__((ext_vector_type(8)))  float  v8f;
typedef __attribute__((ext_vector_type(4)))  float  v4f;
typedef __attribute__((ext_vector_type(4)))  int    v4i;

// addrspace-qualified element types for the async-to-LDS builtin
typedef __attribute__((address_space(1))) v4i v4i_gas;   // global
typedef __attribute__((address_space(3))) v4i v4i_las;   // LDS

#define D_MODEL  1024
#define NUM_HEADS  16
#define D_K        64
#define SEQ      2048
#define BATCH       4
#define MROWS   (BATCH*SEQ)      /* 8192 */

#if __has_builtin(__builtin_amdgcn_global_load_async_to_lds_b128)
#define HAVE_ASYNC_LDS 1
#else
#define HAVE_ASYNC_LDS 0
#endif

// ---------------------------------------------------------------- helpers ---
__device__ __forceinline__ v8f zero8() {
    v8f v;
#pragma unroll
    for (int i = 0; i < 8; i++) v[i] = 0.0f;
    return v;
}
__device__ __forceinline__ v8f splat8(float x) {
    v8f v;
#pragma unroll
    for (int i = 0; i < 8; i++) v[i] = x;
    return v;
}

// Async 16-byte global -> LDS copy (ASYNCcnt path); plain LDS store fallback.
__device__ __forceinline__ void cp_async16(void* lds, const void* g) {
#if HAVE_ASYNC_LDS
    __builtin_amdgcn_global_load_async_to_lds_b128(
        (v4i_gas*)(uintptr_t)g,
        (v4i_las*)(uint32_t)(uintptr_t)lds, 0, 0);
#else
    *(v8bf*)lds = *(const v8bf*)g;
#endif
}
__device__ __forceinline__ void wait_async_all() {
#if HAVE_ASYNC_LDS
#if __has_builtin(__builtin_amdgcn_s_wait_asynccnt)
    __builtin_amdgcn_s_wait_asynccnt(0);
#else
    asm volatile("s_wait_asynccnt 0x0" ::: "memory");
#endif
#endif
}

// A-operand (16x32 bf16): lane m = lane&15, half = lane>>4.
// element e -> k = (e>>3)*16 + (e&7) + half*8   (two 16B chunks)
__device__ __forceinline__ v16bf load_a32k(const bf16* p, int half) {
    v8bf c0 = *(const v8bf*)(p + half * 8);
    v8bf c1 = *(const v8bf*)(p + 16 + half * 8);
    v16bf r;
#pragma unroll
    for (int i = 0; i < 8; i++) { r[i] = c0[i]; r[i + 8] = c1[i]; }
    return r;
}

// B-operand (32x16 bf16): lane n = lane&15, element e -> k = (e>>3)*16 + (e&7) + half*8
__device__ __forceinline__ v16bf load_b32k(const bf16* p, int half) {
    p += half * 8;
    v8bf c0 = *(const v8bf*)(p);
    v8bf c1 = *(const v8bf*)(p + 16);
    v16bf r;
#pragma unroll
    for (int i = 0; i < 8; i++) { r[i] = c0[i]; r[i + 8] = c1[i]; }
    return r;
}

__device__ __forceinline__ v8f wmma_bf16(v16bf a, v16bf b, v8f c) {
    return __builtin_amdgcn_wmma_f32_16x16x32_bf16(
        false, a, false, b, (short)0, c, false, false);
}

// Row reductions across the 16-lane half-group (C-layout rows).
__device__ __forceinline__ v8f red16_max(v8f v) {
#pragma unroll
    for (int mask = 1; mask < 16; mask <<= 1) {
#pragma unroll
        for (int i = 0; i < 8; i++) {
            float o = __shfl_xor(v[i], mask, 32);
            v[i] = fmaxf(v[i], o);
        }
    }
    return v;
}
__device__ __forceinline__ v8f red16_sum(v8f v) {
#pragma unroll
    for (int mask = 1; mask < 16; mask <<= 1) {
#pragma unroll
        for (int i = 0; i < 8; i++) v[i] += __shfl_xor(v[i], mask, 32);
    }
    return v;
}

// ------------------------------------------------------ conversion kernels ---
__global__ __launch_bounds__(256) void f32_to_bf16_kernel(const float* __restrict__ src,
                                                          bf16* __restrict__ dst, int n8) {
    int i = blockIdx.x * blockDim.x + threadIdx.x;
    if (i < n8) {
        v4f a = *(const v4f*)(src + (size_t)i * 8);
        v4f b = *(const v4f*)(src + (size_t)i * 8 + 4);
        v8bf o;
#pragma unroll
        for (int j = 0; j < 4; j++) { o[j] = (bf16)a[j]; o[4 + j] = (bf16)b[j]; }
        *(v8bf*)(dst + (size_t)i * 8) = o;
    }
}

// Wt[n*D + k] = (bf16) W[k*D + n], tiled through LDS.
__global__ __launch_bounds__(256) void transpose_w_bf16_kernel(const float* __restrict__ src,
                                                               bf16* __restrict__ dst) {
    __shared__ float tile[32][33];
    const int bx = blockIdx.x * 32, by = blockIdx.y * 32;
    const int tx = threadIdx.x, ty = threadIdx.y;   // block (32,8)
#pragma unroll
    for (int i = ty; i < 32; i += 8)
        tile[i][tx] = src[(size_t)(by + i) * D_MODEL + bx + tx];
    __syncthreads();
#pragma unroll
    for (int i = ty; i < 32; i += 8)
        dst[(size_t)(bx + i) * D_MODEL + by + tx] = (bf16)tile[tx][i];
}

// ------------------------------------------------------------ GEMM kernel ---
// C[M,N] = A[M,K] * Wt[N,K]^T + bias, double-buffered async LDS pipeline.
// Block tile 128x128, BK=64; 8 waves; wave = 32x64 (2x4 accumulators).
// mode: 0 = bf16 [B,H,S,dk]; 1 = bf16 [B,H,dk,S]; 2 = f32 flat [M,D]
#define BKT 64
#define LDT 72   /* padded LDS row stride (bf16 elems) */

__global__ __launch_bounds__(256) void gemm_bf16_wmma_kernel(
    const bf16* __restrict__ A, const bf16* __restrict__ Wt,
    const float* __restrict__ bias, void* __restrict__ out, int mode) {
    __shared__ bf16 Asm[2][128 * LDT];
    __shared__ bf16 Bsm[2][128 * LDT];
    const int tid  = threadIdx.x;
    const int wave = tid >> 5, lane = tid & 31;
    const int half = lane >> 4, nn = lane & 15;
    const int wm = wave & 3, wn = wave >> 2;       // 4 x 2 wave grid
    const int gm0 = blockIdx.y * 128;
    const int gn0 = blockIdx.x * 128;

    v8f acc[2][4];
#pragma unroll
    for (int s = 0; s < 2; s++)
#pragma unroll
        for (int t = 0; t < 4; t++) acc[s][t] = zero8();

    auto stage = [&](int buf, int kb) {
#pragma unroll
        for (int t = 0; t < 4; t++) {
            const int c = tid + t * 256;           // 0..1023 chunk id
            const int row = c >> 3, col = (c & 7) * 8;
            cp_async16(&Asm[buf][row * LDT + col],
                       A + (size_t)(gm0 + row) * D_MODEL + kb + col);
            cp_async16(&Bsm[buf][row * LDT + col],
                       Wt + (size_t)(gn0 + row) * D_MODEL + kb + col);
        }
    };

    stage(0, 0);
    for (int s = 0; s < D_MODEL / BKT; s++) {      // 16 stages
        const int cur = s & 1;
        wait_async_all();                          // own async writes landed
        __syncthreads();                           // everyone's tiles visible
        if (s + 1 < D_MODEL / BKT) stage(1 - cur, (s + 1) * BKT);
#pragma unroll
        for (int kk = 0; kk < BKT; kk += 32) {
            v16bf af[2];
#pragma unroll
            for (int sub = 0; sub < 2; sub++)
                af[sub] = load_a32k(&Asm[cur][(wm * 32 + sub * 16 + nn) * LDT + kk], half);
#pragma unroll
            for (int t = 0; t < 4; t++) {
                v16bf bm = load_b32k(&Bsm[cur][(wn * 64 + t * 16 + nn) * LDT + kk], half);
#pragma unroll
                for (int sub = 0; sub < 2; sub++)
                    acc[sub][t] = wmma_bf16(af[sub], bm, acc[sub][t]);
            }
        }
    }

#pragma unroll
    for (int sub = 0; sub < 2; sub++) {
#pragma unroll
        for (int t = 0; t < 4; t++) {
            const int   c  = gn0 + wn * 64 + t * 16 + nn;
            const float bv = bias[c];
#pragma unroll
            for (int r = 0; r < 8; r++) {
                const int   g   = gm0 + wm * 32 + sub * 16 + r + half * 8;
                const float val = acc[sub][t][r] + bv;
                if (mode == 2) {
                    ((float*)out)[(size_t)g * D_MODEL + c] = val;
                } else {
                    const int b = g >> 11, sq = g & (SEQ - 1);
                    const int h = c >> 6,  d  = c & (D_K - 1);
                    bf16* ob = (bf16*)out;
                    if (mode == 0)
                        ob[(((size_t)(b * NUM_HEADS + h)) * SEQ + sq) * D_K + d] = (bf16)val;
                    else
                        ob[(((size_t)(b * NUM_HEADS + h)) * D_K + d) * SEQ + sq] = (bf16)val;
                }
            }
        }
    }
}

// -------------------------------------------------------- flash attention ---
// Grid (B*H, S/128), block 256 (8 waves), wave = 16 query rows.
// K (32x64) and Vt (64x32) tiles per 32-key step are async-staged into
// double-buffered LDS once per block; online softmax in registers; P tile
// re-striped C->A layout through a per-wave LDS slab.
__global__ __launch_bounds__(256) void flash_attn_wmma_kernel(
    const bf16* __restrict__ Q, const bf16* __restrict__ K,
    const bf16* __restrict__ Vt, bf16* __restrict__ O) {
    __shared__ bf16 Ksm[2][32 * 72];   // keys x feat, stride 72 (144B, 16B-aligned)
    __shared__ bf16 Vsm[2][64 * 40];   // feat x keys, stride 40 (80B, 16B-aligned)
    __shared__ bf16 Pl[8 * 512];
    const int tid  = threadIdx.x;
    const int wave = tid >> 5, lane = tid & 31;
    const int half = lane >> 4, nn = lane & 15;
    const int bh = blockIdx.x;                       // b*16 + h
    const int q0 = blockIdx.y * 128 + wave * 16;

    const bf16* qrow = Q + ((size_t)bh * SEQ + q0 + nn) * D_K;
    v16bf Qa0 = load_a32k(qrow, half);
    v16bf Qa1 = load_a32k(qrow + 32, half);

    const bf16* Kbh = K + (size_t)bh * SEQ * D_K;    // [S, dk]
    const bf16* Vbh = Vt + (size_t)bh * D_K * SEQ;   // [dk, S]
    bf16* Pw = &Pl[wave * 512];

    v8f o[4];
#pragma unroll
    for (int t = 0; t < 4; t++) o[t] = zero8();
    v8f mrow = splat8(-1e30f);
    v8f lrow = zero8();

    auto stageKV = [&](int buf, int j0) {
        {   // K tile: 32 rows x 64 feats = 256 16B chunks, 1/thread
            const int row = tid >> 3, col = (tid & 7) * 8;
            cp_async16(&Ksm[buf][row * 72 + col],
                       Kbh + (size_t)(j0 + row) * D_K + col);
        }
        {   // V tile: 64 rows(feat) x 32 keys = 256 16B chunks, 1/thread
            const int row = tid >> 2, col = (tid & 3) * 8;
            cp_async16(&Vsm[buf][row * 40 + col],
                       Vbh + (size_t)row * SEQ + j0 + col);
        }
    };

    stageKV(0, 0);
    for (int it = 0; it < SEQ / 32; it++) {
        const int cur = it & 1;
        wait_async_all();
        __syncthreads();
        if (it + 1 < SEQ / 32) stageKV(1 - cur, (it + 1) * 32);

        // ---- scores: two 16x16 tiles over 32 keys, K from LDS ----
        v8f s0 = zero8(), s1 = zero8();
        s0 = wmma_bf16(Qa0, load_b32k(&Ksm[cur][nn * 72 + 0], half), s0);
        s0 = wmma_bf16(Qa1, load_b32k(&Ksm[cur][nn * 72 + 32], half), s0);
        s1 = wmma_bf16(Qa0, load_b32k(&Ksm[cur][(16 + nn) * 72 + 0], half), s1);
        s1 = wmma_bf16(Qa1, load_b32k(&Ksm[cur][(16 + nn) * 72 + 32], half), s1);

        // ---- online softmax ----
        v8f t;
#pragma unroll
        for (int i = 0; i < 8; i++) {
            s0[i] *= 0.125f; s1[i] *= 0.125f;
            t[i] = fmaxf(s0[i], s1[i]);
        }
        t = red16_max(t);
        v8f mnew, alpha, p0, p1, rs;
#pragma unroll
        for (int i = 0; i < 8; i++) {
            mnew[i]  = fmaxf(mrow[i], t[i]);
            alpha[i] = __expf(mrow[i] - mnew[i]);
            p0[i]    = __expf(s0[i] - mnew[i]);
            p1[i]    = __expf(s1[i] - mnew[i]);
            rs[i]    = p0[i] + p1[i];
        }
        rs = red16_sum(rs);
#pragma unroll
        for (int i = 0; i < 8; i++) {
            lrow[i] = lrow[i] * alpha[i] + rs[i];
            mrow[i] = mnew[i];
        }
#pragma unroll
        for (int tt = 0; tt < 4; tt++)
#pragma unroll
            for (int i = 0; i < 8; i++) o[tt][i] *= alpha[i];

        // ---- P: C-layout -> A-layout via per-wave LDS slab ----
#pragma unroll
        for (int r = 0; r < 8; r++) {
            Pw[(r + half * 8) * 32 + nn]      = (bf16)p0[r];
            Pw[(r + half * 8) * 32 + 16 + nn] = (bf16)p1[r];
        }
        asm volatile("s_wait_dscnt 0" ::: "memory");   // same-wave store->load
        v16bf Pa = load_a32k(&Pw[nn * 32], half);

        // ---- O += P @ V, V from LDS ----
#pragma unroll
        for (int tt = 0; tt < 4; tt++)
            o[tt] = wmma_bf16(Pa, load_b32k(&Vsm[cur][(tt * 16 + nn) * 40], half), o[tt]);
    }

    // ---- epilogue: normalize, scatter into [B, S, D] bf16 ----
    const int b = bh >> 4, h = bh & 15;
#pragma unroll
    for (int r = 0; r < 8; r++) {
        const int   g   = q0 + r + half * 8;
        const float inv = 1.0f / lrow[r];
        bf16* ob = O + ((size_t)b * SEQ + g) * D_MODEL + h * D_K;
#pragma unroll
        for (int tt = 0; tt < 4; tt++)
            ob[tt * 16 + nn] = (bf16)(o[tt][r] * inv);
    }
}

// ---------------------------------------------------------------- launcher ---
extern "C" void kernel_launch(void* const* d_in, const int* in_sizes, int n_in,
                              void* d_out, int out_size, void* d_ws, size_t ws_size,
                              hipStream_t stream) {
    (void)in_sizes; (void)n_in; (void)out_size; (void)ws_size;
    const float* query = (const float*)d_in[0];
    const float* key_  = (const float*)d_in[1];
    const float* value = (const float*)d_in[2];
    const float* wq = (const float*)d_in[3];  const float* bq = (const float*)d_in[4];
    const float* wk = (const float*)d_in[5];  const float* bk = (const float*)d_in[6];
    const float* wv = (const float*)d_in[7];  const float* bv = (const float*)d_in[8];
    const float* wo = (const float*)d_in[9];  const float* bo = (const float*)d_in[10];

    const size_t MD = (size_t)MROWS * D_MODEL;
    const size_t DD = (size_t)D_MODEL * D_MODEL;
    char* w = (char*)d_ws;
    auto carve = [&](size_t bytes) {
        void* p = (void*)w;
        w += (bytes + 255) & ~(size_t)255;
        return p;
    };
    bf16* Xq  = (bf16*)carve(MD * 2);
    bf16* Xk  = (bf16*)carve(MD * 2);
    bf16* Xv  = (bf16*)carve(MD * 2);
    bf16* Wqt = (bf16*)carve(DD * 2);
    bf16* Wkt = (bf16*)carve(DD * 2);
    bf16* Wvt = (bf16*)carve(DD * 2);
    bf16* Wot = (bf16*)carve(DD * 2);
    bf16* Qb  = (bf16*)carve(MD * 2);
    bf16* Kb  = (bf16*)carve(MD * 2);
    bf16* Vtb = (bf16*)carve(MD * 2);
    bf16* Ob  = (bf16*)carve(MD * 2);

    // 1) activations fp32 -> bf16 (vectorized b128)
    const int n8 = (int)(MD / 8);
    dim3 cblk(256), cgrd((n8 + 255) / 256);
    f32_to_bf16_kernel<<<cgrd, cblk, 0, stream>>>(query, Xq, n8);
    f32_to_bf16_kernel<<<cgrd, cblk, 0, stream>>>(key_,  Xk, n8);
    f32_to_bf16_kernel<<<cgrd, cblk, 0, stream>>>(value, Xv, n8);

    // 2) weights: transpose + convert
    dim3 tblk(32, 8), tgrd(D_MODEL / 32, D_MODEL / 32);
    transpose_w_bf16_kernel<<<tgrd, tblk, 0, stream>>>(wq, Wqt);
    transpose_w_bf16_kernel<<<tgrd, tblk, 0, stream>>>(wk, Wkt);
    transpose_w_bf16_kernel<<<tgrd, tblk, 0, stream>>>(wv, Wvt);
    transpose_w_bf16_kernel<<<tgrd, tblk, 0, stream>>>(wo, Wot);

    // 3) projections
    dim3 gblk(256), ggrd(D_MODEL / 128, MROWS / 128);
    gemm_bf16_wmma_kernel<<<ggrd, gblk, 0, stream>>>(Xq, Wqt, bq, (void*)Qb,  0);
    gemm_bf16_wmma_kernel<<<ggrd, gblk, 0, stream>>>(Xk, Wkt, bk, (void*)Kb,  0);
    gemm_bf16_wmma_kernel<<<ggrd, gblk, 0, stream>>>(Xv, Wvt, bv, (void*)Vtb, 1);

    // 4) flash attention
    dim3 fblk(256), fgrd(BATCH * NUM_HEADS, SEQ / 128);
    flash_attn_wmma_kernel<<<fgrd, fblk, 0, stream>>>(Qb, Kb, Vtb, Ob);

    // 5) output projection -> f32
    gemm_bf16_wmma_kernel<<<ggrd, gblk, 0, stream>>>(Ob, Wot, bo, d_out, 2);
}
